// PaiNNModel_2791728742504
// MI455X (gfx1250) — matrix operand order repaired and checked by reading, not compiled, not run on hardware
//
#include <hip/hip_runtime.h>
#include <hip/hip_bf16.h>
#include <math.h>

// ---------------------------------------------------------------------------
// PaiNN forward on gfx1250: WMMA f16->f32 GEMMs for all dense layers
// (weights prepacked to f16), fused bandwidth/atomic kernel for the edge
// message passing.
// ---------------------------------------------------------------------------

#define PAINN_N 20000
#define PAINN_E 320000
#define PAINN_F 128
#define PAINN_RBF 20
#define PAINN_RC 5.0f
#define PAINN_NM 64

typedef __attribute__((ext_vector_type(16))) _Float16 v16h;
typedef __attribute__((ext_vector_type(8)))  float    v8f;

__device__ __forceinline__ float fast_silu(float x) {
  // x * sigmoid(x) with v_rcp_f32 instead of the IEEE divide expansion
  return x * __builtin_amdgcn_rcpf(1.0f + __expf(-x));
}

// ---------------------------------------------------------------------------
// WMMA GEMM: OUT(nrows x M) = act( X(nrows x K) @ W(M x K)^T + bias )
// W is prepacked f16 (row-major M x K). Each wave owns RT 16-row tiles; A
// fragments live in registers across all column tiles, each B fragment feeds
// RT WMMAs.
//
// PRECONDITION: nrows % (16*RT) == 0 (true here: 20000 and 60000 rows with
// RT=2 -> 625/1875 wave-pairs). No per-tile tail guard, so there is no
// divergent predication around v_wmma and EXEC stays all-ones (required by
// ISA 7.12); the only exit is the wave-uniform tile0 bound check.
//
// Layouts used here (wave32; A and B share a contiguous k order):
//   A frag (16x32 f16): lane l -> row l%16, K = kt*32 + (l/16)*16 + 0..15
//   B frag (32x16 f16): lane l -> col l%16, K = kt*32 + (l/16)*16 + 0..15
//   D (16x16 f32):      elem r of lane l -> row r + 8*(l/16), col l%16
// ---------------------------------------------------------------------------
template <int K, int M, int RT, bool SILU, bool BIAS>
__global__ __launch_bounds__(128) void wmma_gemm(
    const float* __restrict__ X, const _Float16* __restrict__ W,
    const float* __restrict__ B, float* __restrict__ OUT, int nrows) {
  constexpr int NKT = K / 32;   // k-steps
  constexpr int NCT = M / 16;   // column tiles
  const int wave = threadIdx.x >> 5;
  const int lane = threadIdx.x & 31;
  const int ntiles = nrows >> 4;
  const int tile0 = (blockIdx.x * 4 + wave) * RT;
  if (tile0 >= ntiles) return;                       // wave-uniform exit
  const int mrow = lane & 15;
  const int hi   = lane >> 4;   // lane half selects K sub-block

  // Load A fragments (RT row tiles, full K), convert fp32 -> f16 once.
  v16h A[RT][NKT];
#pragma unroll
  for (int t = 0; t < RT; ++t) {
    const float* xrow = X + (size_t)((tile0 + t) * 16 + mrow) * K;
#pragma unroll
    for (int kt = 0; kt < NKT; ++kt) {
      const float* p = xrow + kt * 32 + hi * 16;
#pragma unroll
      for (int j = 0; j < 8; ++j) {
        A[t][kt][j]     = (_Float16)p[j];        // K = base + 0..7
        A[t][kt][j + 8] = (_Float16)p[j + 8];    // K = base + 8..15
      }
    }
  }

#pragma unroll 1
  for (int ct = 0; ct < NCT; ++ct) {
    const int col = ct * 16 + mrow;
    const _Float16* wrow = W + (size_t)col * K + hi * 16;
    if (ct + 1 < NCT)
      __builtin_prefetch(wrow + (size_t)16 * K, 0, 1);  // next col-tile weights
    v8f acc[RT];
#pragma unroll
    for (int t = 0; t < RT; ++t) acc[t] = (v8f){};
#pragma unroll
    for (int kt = 0; kt < NKT; ++kt) {
      const v16h Bf = *(const v16h*)(wrow + kt * 32);   // 16 contiguous f16
#pragma unroll
      for (int t = 0; t < RT; ++t) {
        acc[t] = __builtin_amdgcn_wmma_f32_16x16x32_f16(
            false, A[t][kt], false, Bf, (short)0, acc[t], false, false);
      }
    }
    const float b = BIAS ? B[col] : 0.0f;
#pragma unroll
    for (int t = 0; t < RT; ++t) {
      const int row0 = (tile0 + t) * 16;
#pragma unroll
      for (int rr = 0; rr < 8; ++rr) {
        float val = acc[t][rr] + b;
        if (SILU) val = fast_silu(val);
        OUT[(size_t)(row0 + rr + hi * 8) * M + col] = val;
      }
    }
  }
}

// ---------------------------------------------------------------------------
// Weight prepack: fp32 -> f16 (layout unchanged, row-major M x K)
// ---------------------------------------------------------------------------
__global__ void cvt_f16_kernel(const float* __restrict__ src,
                               _Float16* __restrict__ dst, int n) {
  const int i = blockIdx.x * blockDim.x + threadIdx.x;
  if (i < n) dst[i] = (_Float16)src[i];
}

// ---------------------------------------------------------------------------
// init: s = emb[z], v = 0, out = 0
// ---------------------------------------------------------------------------
__global__ void init_kernel(const int* __restrict__ z, const float* __restrict__ emb,
                            float* __restrict__ s, float* __restrict__ v,
                            float* __restrict__ out) {
  const int i = blockIdx.x * blockDim.x + threadIdx.x;
  if (i < PAINN_NM) out[i] = 0.0f;
  if (i < PAINN_N * PAINN_F) {
    const int node = i >> 7, f = i & 127;
    s[i] = emb[z[node] * PAINN_F + f];
  }
  if (i < PAINN_N * 3 * PAINN_F) v[i] = 0.0f;
}

// ---------------------------------------------------------------------------
// RBF expansion + cosine cutoff (iteration-invariant)
// ---------------------------------------------------------------------------
__global__ void rbf_kernel(const float* __restrict__ dist, float* __restrict__ rbfb,
                           float* __restrict__ cutb) {
  const int e = blockIdx.x * blockDim.x + threadIdx.x;
  if (e >= PAINN_E) return;
  const float d = dist[e];
  const float inv = __builtin_amdgcn_rcpf(d);
  const float PI = 3.14159265358979f;
#pragma unroll
  for (int k = 0; k < PAINN_RBF; ++k)
    rbfb[e * PAINN_RBF + k] = __sinf((float)(k + 1) * PI * d / PAINN_RC) * inv;
  cutb[e] = (d < PAINN_RC) ? 0.5f * (__cosf(PI * d / PAINN_RC) + 1.0f) : 0.0f;
}

// ---------------------------------------------------------------------------
// Fused edge kernel: dist_rep = (rbf@ew^T + eb)*cut, res = a[dst]*dist_rep,
// scatter 2*rs into s[src], 2*(v[dst]*rv + sense*rd) into v_out[src].
// ew/eb staged in LDS, reused across 8 edges per block. Thread = feature f.
// ---------------------------------------------------------------------------
#define EDGES_PER_BLOCK 8
__global__ __launch_bounds__(128) void edge_kernel(
    const int* __restrict__ graph, const float* __restrict__ rbfb,
    const float* __restrict__ cutb, const float* __restrict__ sense,
    const float* __restrict__ a, const float* __restrict__ v_in,
    float* __restrict__ s_io, float* __restrict__ v_out,
    const float* __restrict__ ew, const float* __restrict__ eb) {
  __shared__ float sew[3 * PAINN_F * PAINN_RBF];
  __shared__ float seb[3 * PAINN_F];
  for (int i = threadIdx.x; i < 3 * PAINN_F * PAINN_RBF; i += 128) sew[i] = ew[i];
  for (int i = threadIdx.x; i < 3 * PAINN_F; i += 128) seb[i] = eb[i];
  __syncthreads();
  const int f = threadIdx.x;
#pragma unroll 1
  for (int ee = 0; ee < EDGES_PER_BLOCK; ++ee) {
    const int e = blockIdx.x * EDGES_PER_BLOCK + ee;
    const int src = graph[2 * e];
    const int dst = graph[2 * e + 1];
    const float cut = cutb[e];
    float rb[PAINN_RBF];
#pragma unroll
    for (int i = 0; i < PAINN_RBF; ++i) rb[i] = rbfb[e * PAINN_RBF + i];
    float drv = seb[f], drs = seb[PAINN_F + f], drd = seb[2 * PAINN_F + f];
#pragma unroll
    for (int i = 0; i < PAINN_RBF; ++i) {
      drv += rb[i] * sew[f * PAINN_RBF + i];
      drs += rb[i] * sew[(PAINN_F + f) * PAINN_RBF + i];
      drd += rb[i] * sew[(2 * PAINN_F + f) * PAINN_RBF + i];
    }
    drv *= cut; drs *= cut; drd *= cut;
    const size_t ab = (size_t)dst * (3 * PAINN_F);
    const float rv = a[ab + f] * drv;
    const float rs = a[ab + PAINN_F + f] * drs;
    const float rd = a[ab + 2 * PAINN_F + f] * drd;
    atomicAdd(&s_io[(size_t)src * PAINN_F + f], 2.0f * rs);
    const float sx = sense[3 * e], sy = sense[3 * e + 1], sz = sense[3 * e + 2];
    const size_t vd = (size_t)dst * (3 * PAINN_F);
    const size_t vs = (size_t)src * (3 * PAINN_F);
    atomicAdd(&v_out[vs + f],               2.0f * (v_in[vd + f] * rv + sx * rd));
    atomicAdd(&v_out[vs + PAINN_F + f],     2.0f * (v_in[vd + PAINN_F + f] * rv + sy * rd));
    atomicAdd(&v_out[vs + 2 * PAINN_F + f], 2.0f * (v_in[vd + 2 * PAINN_F + f] * rv + sz * rd));
  }
}

// ---------------------------------------------------------------------------
// h = [s, ||Vv||_dim]
// ---------------------------------------------------------------------------
__global__ void build_h_kernel(const float* __restrict__ s, const float* __restrict__ Vv,
                               float* __restrict__ h) {
  const int i = blockIdx.x * blockDim.x + threadIdx.x;
  if (i >= PAINN_N * PAINN_F) return;
  const int node = i >> 7, f = i & 127;
  const size_t b = (size_t)node * (3 * PAINN_F);
  const float a0 = Vv[b + f], a1 = Vv[b + PAINN_F + f], a2 = Vv[b + 2 * PAINN_F + f];
  h[(size_t)node * 256 + f] = s[i];
  h[(size_t)node * 256 + PAINN_F + f] = sqrtf(a0 * a0 + a1 * a1 + a2 * a2);
}

// ---------------------------------------------------------------------------
// Gated update: s += ass + asv * <Uv,Vv>_dim ; v = vn + avv * Uv
// ---------------------------------------------------------------------------
__global__ void update_kernel(float* __restrict__ s, float* __restrict__ v,
                              const float* __restrict__ vn, const float* __restrict__ Uv,
                              const float* __restrict__ Vv, const float* __restrict__ a2) {
  const int i = blockIdx.x * blockDim.x + threadIdx.x;
  if (i >= PAINN_N * PAINN_F) return;
  const int node = i >> 7, f = i & 127;
  const size_t b = (size_t)node * (3 * PAINN_F);
  const float u0 = Uv[b + f], u1 = Uv[b + PAINN_F + f], u2 = Uv[b + 2 * PAINN_F + f];
  const float w0 = Vv[b + f], w1 = Vv[b + PAINN_F + f], w2 = Vv[b + 2 * PAINN_F + f];
  const float dot = u0 * w0 + u1 * w1 + u2 * w2;
  const float avv = a2[b + f];
  const float asv = a2[b + PAINN_F + f];
  const float ass = a2[b + 2 * PAINN_F + f];
  s[i] += ass + asv * dot;
  v[b + f]               = vn[b + f] + avv * u0;
  v[b + PAINN_F + f]     = vn[b + PAINN_F + f] + avv * u1;
  v[b + 2 * PAINN_F + f] = vn[b + 2 * PAINN_F + f] + avv * u2;
}

// ---------------------------------------------------------------------------
// Final head: out = h1 @ ow2^T + ob2, atomic segment-sum by graph_idx
// ---------------------------------------------------------------------------
__global__ void final_kernel(const float* __restrict__ h1, const float* __restrict__ ow2,
                             const float* __restrict__ ob2, const int* __restrict__ gidx,
                             float* __restrict__ out) {
  const int node = blockIdx.x * blockDim.x + threadIdx.x;
  if (node >= PAINN_N) return;
  float acc = ob2[0];
#pragma unroll 4
  for (int k = 0; k < PAINN_F; ++k) acc += h1[(size_t)node * PAINN_F + k] * ow2[k];
  atomicAdd(&out[gidx[node]], acc);
}

// ---------------------------------------------------------------------------
// Host orchestration
// ---------------------------------------------------------------------------
extern "C" void kernel_launch(void* const* d_in, const int* in_sizes, int n_in,
                              void* d_out, int out_size, void* d_ws, size_t ws_size,
                              hipStream_t stream) {
  const int    N = PAINN_N, E = PAINN_E, F = PAINN_F;
  const int*   z      = (const int*)d_in[0];
  const int*   graph  = (const int*)d_in[1];
  const float* dist   = (const float*)d_in[2];
  const float* sense  = (const float*)d_in[3];
  const int*   gidx   = (const int*)d_in[4];
  const float* emb    = (const float*)d_in[5];
  const float* mw1    = (const float*)d_in[6];
  const float* mb1    = (const float*)d_in[7];
  const float* mw2    = (const float*)d_in[8];
  const float* mb2    = (const float*)d_in[9];
  const float* ew     = (const float*)d_in[10];
  const float* eb     = (const float*)d_in[11];
  const float* uw     = (const float*)d_in[12];
  const float* vw     = (const float*)d_in[13];
  const float* aw1    = (const float*)d_in[14];
  const float* ab1    = (const float*)d_in[15];
  const float* aw2    = (const float*)d_in[16];
  const float* ab2    = (const float*)d_in[17];
  const float* ow1    = (const float*)d_in[18];
  const float* ob1    = (const float*)d_in[19];
  const float* ow2    = (const float*)d_in[20];
  const float* ob2    = (const float*)d_in[21];
  float* out = (float*)d_out;

  // Workspace carve-up: fp32 buffers first, then f16 weight packs
  float* ws = (float*)d_ws;
  size_t off = 0;
  float* s    = ws + off; off += (size_t)N * F;        // node scalars
  float* v    = ws + off; off += (size_t)N * 3 * F;    // node vectors (current)
  float* vn   = ws + off; off += (size_t)N * 3 * F;    // node vectors (next)
  float* a    = ws + off; off += (size_t)N * 3 * F;    // 3F activations (a / a2)
  float* h1   = ws + off; off += (size_t)N * F;        // MLP hidden
  float* Uv   = ws + off; off += (size_t)N * 3 * F;
  float* Vv   = ws + off; off += (size_t)N * 3 * F;
  float* h    = ws + off; off += (size_t)N * 2 * F;    // [s, |Vv|]
  float* rbfb = ws + off; off += (size_t)E * PAINN_RBF;
  float* cutb = ws + off; off += (size_t)E;
  _Float16* wh = (_Float16*)(ws + off);
  size_t hoff = 0;
  _Float16* wmw1 = wh + hoff; hoff += (size_t)3 * F * F;
  _Float16* wmw2 = wh + hoff; hoff += (size_t)3 * 3 * F * F;
  _Float16* wuw  = wh + hoff; hoff += (size_t)3 * F * F;
  _Float16* wvw  = wh + hoff; hoff += (size_t)3 * F * F;
  _Float16* waw1 = wh + hoff; hoff += (size_t)3 * F * 2 * F;
  _Float16* waw2 = wh + hoff; hoff += (size_t)3 * 3 * F * F;
  _Float16* wow1 = wh + hoff; hoff += (size_t)F * F;

  // ---- weight prepack (fp32 -> f16), all iterations at once ----
  {
    const int n1 = 3 * F * F, n2 = 3 * 3 * F * F, n3 = 3 * F * 2 * F, n4 = F * F;
    cvt_f16_kernel<<<(n1 + 255) / 256, 256, 0, stream>>>(mw1, wmw1, n1);
    cvt_f16_kernel<<<(n2 + 255) / 256, 256, 0, stream>>>(mw2, wmw2, n2);
    cvt_f16_kernel<<<(n1 + 255) / 256, 256, 0, stream>>>(uw,  wuw,  n1);
    cvt_f16_kernel<<<(n1 + 255) / 256, 256, 0, stream>>>(vw,  wvw,  n1);
    cvt_f16_kernel<<<(n3 + 255) / 256, 256, 0, stream>>>(aw1, waw1, n3);
    cvt_f16_kernel<<<(n2 + 255) / 256, 256, 0, stream>>>(aw2, waw2, n2);
    cvt_f16_kernel<<<(n4 + 255) / 256, 256, 0, stream>>>(ow1, wow1, n4);
  }

  // GEMM grids: blocks consume 4 waves x RT tiles
  const int tN  = N / 16;                       // 1250 row tiles (even)
  const int t3N = 3 * N / 16;                   // 3750 row tiles (even)
  const int gN2  = (tN / 2 + 3) / 4;            // RT=2: 625 wave-pairs
  const int g3N2 = (t3N / 2 + 3) / 4;           // RT=2: 1875 wave-pairs
  const int gN1  = (tN + 3) / 4;                // RT=1 (K=256 GEMM)

  init_kernel<<<(N * 3 * F + 255) / 256, 256, 0, stream>>>(z, emb, s, v, out);
  rbf_kernel<<<(E + 255) / 256, 256, 0, stream>>>(dist, rbfb, cutb);

  for (int i = 0; i < 3; ++i) {
    const _Float16* mw1i = wmw1 + (size_t)i * F * F;
    const float*    mb1i = mb1 + (size_t)i * F;
    const _Float16* mw2i = wmw2 + (size_t)i * 3 * F * F;
    const float*    mb2i = mb2 + (size_t)i * 3 * F;
    const float*    ewi  = ew  + (size_t)i * 3 * F * PAINN_RBF;
    const float*    ebi  = eb  + (size_t)i * 3 * F;
    const _Float16* uwi  = wuw + (size_t)i * F * F;
    const _Float16* vwi  = wvw + (size_t)i * F * F;
    const _Float16* aw1i = waw1 + (size_t)i * F * 2 * F;
    const float*    ab1i = ab1 + (size_t)i * F;
    const _Float16* aw2i = waw2 + (size_t)i * 3 * F * F;
    const float*    ab2i = ab2 + (size_t)i * 3 * F;

    // a = silu(s@mw1^T + mb1) @ mw2^T + mb2
    wmma_gemm<128, 128, 2, true,  true><<<gN2, 128, 0, stream>>>(s,  mw1i, mb1i, h1, N);
    wmma_gemm<128, 384, 2, false, true><<<gN2, 128, 0, stream>>>(h1, mw2i, mb2i, a,  N);

    // double-buffer v: edge kernel reads v, accumulates into vn
    hipMemcpyAsync(vn, v, (size_t)N * 3 * F * sizeof(float),
                   hipMemcpyDeviceToDevice, stream);
    edge_kernel<<<E / EDGES_PER_BLOCK, 128, 0, stream>>>(
        graph, rbfb, cutb, sense, a, v, s, vn, ewi, ebi);

    // Uv = vn @ uw^T, Vv = vn @ vw^T  (3N rows of F)
    wmma_gemm<128, 128, 2, false, false><<<g3N2, 128, 0, stream>>>(vn, uwi, nullptr, Uv, 3 * N);
    wmma_gemm<128, 128, 2, false, false><<<g3N2, 128, 0, stream>>>(vn, vwi, nullptr, Vv, 3 * N);

    build_h_kernel<<<(N * F + 255) / 256, 256, 0, stream>>>(s, Vv, h);

    // a2 = silu(h@aw1^T + ab1) @ aw2^T + ab2
    wmma_gemm<256, 128, 1, true,  true><<<gN1, 128, 0, stream>>>(h,  aw1i, ab1i, h1, N);
    wmma_gemm<128, 384, 2, false, true><<<gN2, 128, 0, stream>>>(h1, aw2i, ab2i, a,  N);

    update_kernel<<<(N * F + 255) / 256, 256, 0, stream>>>(s, v, vn, Uv, Vv, a);
  }

  // out head
  wmma_gemm<128, 128, 2, true, true><<<gN2, 128, 0, stream>>>(s, wow1, ob1, h1, N);
  final_kernel<<<(N + 255) / 256, 256, 0, stream>>>(h1, ow2, ob2, gidx, out);

  (void)in_sizes; (void)n_in; (void)out_size; (void)ws_size;
}
